// Embedding_47699906789800
// MI455X (gfx1250) — matrix operand order repaired and checked by reading, not compiled, not run on hardware
//
#include <hip/hip_runtime.h>
#include <hip/hip_bf16.h>

typedef __attribute__((ext_vector_type(16))) _Float16 v16h;
typedef __attribute__((ext_vector_type(8)))  _Float16 v8h;
typedef __attribute__((ext_vector_type(4)))  _Float16 v4h;
typedef __attribute__((ext_vector_type(8)))  float    v8f;
typedef __attribute__((ext_vector_type(4)))  unsigned int u32x4;
typedef __attribute__((ext_vector_type(8)))  int i32x8;
typedef __attribute__((ext_vector_type(4)))  int i32x4;

union V16 { v16h v; v8h h[2]; _Float16 e[16]; };
union V8  { v8h  v; _Float16 e[8]; unsigned short u[8]; };

static __device__ __forceinline__ v8f wmma_f16(v16h a, v16h b, v8f c) {
  // 8 args: (neg_a, A, neg_b, B, c_mod, C, reuse_a, reuse_b)
  return __builtin_amdgcn_wmma_f32_16x16x32_f16(false, a, false, b, (short)0, c, false, false);
}

static __device__ __forceinline__ v8f vzero() {
  v8f z = {0.f,0.f,0.f,0.f,0.f,0.f,0.f,0.f};
  return z;
}

// ---------------------------------------------------------------------------
// Tensor Data Mover: 2D tile load Global -> LDS.
// D# per CDNA5 ISA ch.8: group0 = {count, lds_addr, global_addr, type=2},
// group1 = {data_size=2B, tensor dims/strides, tile dims}.
// tile_k elems contiguous per row, tile_rows rows, row stride = stride elems.
// Rows are packed back-to-back in LDS starting at lds_off.
// This toolchain exposes the 6-arg builtin:
//   (u32x4 g0, i32x8 g1, i32x4 g2, i32x4 g3, i32x8 extra, i32 cpol)
// ---------------------------------------------------------------------------
static __device__ __forceinline__ void tdm_load_2d_f16(const _Float16* gptr,
                                                       unsigned lds_off,
                                                       unsigned tile_k,
                                                       unsigned tile_rows,
                                                       unsigned stride_elems,
                                                       unsigned tensor_d0,
                                                       unsigned tensor_d1) {
  unsigned long long ga = (unsigned long long)(const void*)gptr;
  u32x4 g0;
  g0[0] = 1u;                                        // count=1 valid descriptor
  g0[1] = lds_off;                                   // LDS byte address
  g0[2] = (unsigned)ga;                              // global_addr[31:0]
  g0[3] = (unsigned)((ga >> 32) & 0x01FFFFFFull)     // global_addr[56:32]
        | 0x80000000u;                               // type = 2 ("image")
  i32x8 g1;
  g1[0] = (int)(1u << 16);                           // data_size = 1 -> 2 bytes
  g1[1] = (int)((tensor_d0 & 0xFFFFu) << 16);        // tensor_dim0[15:0]
  g1[2] = (int)((tensor_d0 >> 16) | ((tensor_d1 & 0xFFFFu) << 16));
  g1[3] = (int)((tensor_d1 >> 16) | (tile_k << 16)); // tile_dim0
  g1[4] = (int)(tile_rows & 0xFFFFu);                // tile_dim1 (tile_dim2=0)
  g1[5] = (int)stride_elems;                         // tensor_dim0_stride[31:0]
  g1[6] = 0;                                         // stride[47:32], dim1_stride lo
  g1[7] = 0;
  i32x4 gz4;
  gz4[0] = 0; gz4[1] = 0; gz4[2] = 0; gz4[3] = 0;
  i32x8 gz8;
  gz8[0] = 0; gz8[1] = 0; gz8[2] = 0; gz8[3] = 0;
  gz8[4] = 0; gz8[5] = 0; gz8[6] = 0; gz8[7] = 0;
  __builtin_amdgcn_tensor_load_to_lds(g0, g1, gz4, gz4, gz8, 0);
}

// ---------------------------------------------------------------------------
// fp32 -> f16 conversion (vectorized x4)
// ---------------------------------------------------------------------------
__global__ __launch_bounds__(256) void convert_f32_f16(const float* __restrict__ src,
                                                       _Float16* __restrict__ dst, int n4) {
  int i = blockIdx.x * 256 + threadIdx.x;
  if (i < n4) {
    float4 f = ((const float4*)src)[i];
    v4h o;
    o[0] = (_Float16)f.x; o[1] = (_Float16)f.y; o[2] = (_Float16)f.z; o[3] = (_Float16)f.w;
    *(v4h*)(dst + (size_t)i * 4) = o;
  }
}

// ---------------------------------------------------------------------------
// Generic f16 WMMA GEMM: C[M,N] = A[M,K] * B[K,N] (+bias), A,B row-major f16.
// Block 256 thr (8 waves, 2x4 wave grid), tile 128x128, BK=32.
// A tile staged by the Tensor Data Mover (wave 0 issues the DMA);
// B tile transposed into LDS with packed b32 stores.
// ---------------------------------------------------------------------------
template<typename OutT, bool HAS_BIAS>
__global__ __launch_bounds__(256)
void gemm_f16(const _Float16* __restrict__ A, const _Float16* __restrict__ B,
              OutT* __restrict__ C, const float* __restrict__ bias,
              int M, int N, int K, size_t sA, size_t sB, size_t sC) {
  __shared__ __align__(32) _Float16 As[128 * 32];  // [row][k]
  __shared__ __align__(32) _Float16 Bs[128 * 40];  // [n][k], padded stride 40
  unsigned* Bs32 = (unsigned*)Bs;                  // row stride 20 dwords
  const int tid = threadIdx.x, lane = tid & 31, wave = tid >> 5;
  const int wm = wave >> 2, wn = wave & 3;
  const int la = lane & 15, lh = lane >> 4;
  const int m0 = blockIdx.y * 128, n0 = blockIdx.x * 128;
  A += (size_t)blockIdx.z * sA;
  B += (size_t)blockIdx.z * sB;
  C += (size_t)blockIdx.z * sC;

  v8f acc[4][2];
#pragma unroll
  for (int mi = 0; mi < 4; ++mi)
#pragma unroll
    for (int nj = 0; nj < 2; ++nj) acc[mi][nj] = vzero();

  const int kp  = tid & 15;          // k-pair index (covers k = 2*kp, 2*kp+1)
  const int nc8 = (tid >> 4) * 8;    // 8-column chunk within 128
  const unsigned ldsA = (unsigned)(size_t)(void*)As;

  for (int k0 = 0; k0 < K; k0 += 32) {
    // stage A tile [128 x 32] via Tensor Data Mover (one DMA from wave 0)
    if (wave == 0) {
      tdm_load_2d_f16(A + (size_t)m0 * K + k0, ldsA,
                      /*tile_k=*/32, /*tile_rows=*/128,
                      /*stride=*/(unsigned)K,
                      /*tensor_d0=*/(unsigned)K, /*tensor_d1=*/128u);
      __builtin_amdgcn_s_wait_tensorcnt(0);
    }
    // stage B tile [32 x 128] transposed into Bs[n][k] (two k's packed per b32)
    const _Float16* bp0 = B + (size_t)(k0 + 2 * kp) * N + n0 + nc8;
    V8 t0, t1;
    t0.v = *(const v8h*)bp0;
    t1.v = *(const v8h*)(bp0 + N);
    __builtin_prefetch(bp0 + (size_t)32 * N, 0, 1);
#pragma unroll
    for (int e = 0; e < 8; ++e)
      Bs32[(nc8 + e) * 20 + kp] = (unsigned)t0.u[e] | ((unsigned)t1.u[e] << 16);
    __syncthreads();

    V16 a[4], b[2];
#pragma unroll
    for (int mi = 0; mi < 4; ++mi) {
      int r = wm * 64 + mi * 16 + la;
      a[mi].h[0] = *(const v8h*)&As[r * 32 + lh * 16];
      a[mi].h[1] = *(const v8h*)&As[r * 32 + lh * 16 + 8];
    }
#pragma unroll
    for (int nj = 0; nj < 2; ++nj) {
      int n = wn * 32 + nj * 16 + la;
      b[nj].v = *(const v16h*)&Bs[n * 40 + lh * 16];
    }
#pragma unroll
    for (int mi = 0; mi < 4; ++mi)
#pragma unroll
      for (int nj = 0; nj < 2; ++nj)
        acc[mi][nj] = wmma_f16(a[mi].v, b[nj].v, acc[mi][nj]);
    __syncthreads();
  }

#pragma unroll
  for (int mi = 0; mi < 4; ++mi)
#pragma unroll
    for (int nj = 0; nj < 2; ++nj)
#pragma unroll
      for (int r = 0; r < 8; ++r) {
        int row = m0 + wm * 64 + mi * 16 + r + lh * 8;
        int col = n0 + wn * 32 + nj * 16 + la;
        float v = acc[mi][nj][r];
        if (HAS_BIAS) v += bias[row];
        C[(size_t)row * N + col] = (OutT)v;
      }
}

// ---------------------------------------------------------------------------
// Depthwise 3x3 SAME conv on f16 activations, fp32 accumulate.
// Y,Z layout [s][1152][128*128], weights fp32 [1152][9].
// ---------------------------------------------------------------------------
__global__ __launch_bounds__(256)
void dwconv3x3(const _Float16* __restrict__ Y, const float* __restrict__ W,
               _Float16* __restrict__ Z) {
  int n = blockIdx.x * 256 + threadIdx.x;     // 0..16383
  int o = blockIdx.y;                          // 0..1151
  int s = blockIdx.z;                          // 0..7
  int yy = n >> 7, xx = n & 127;
  const _Float16* p = Y + ((size_t)(s * 1152 + o) << 14);
  float acc = 0.f;
#pragma unroll
  for (int ky = 0; ky < 3; ++ky) {
    int yq = yy + ky - 1;
    if ((unsigned)yq >= 128u) continue;
#pragma unroll
    for (int kx = 0; kx < 3; ++kx) {
      int xq = xx + kx - 1;
      if ((unsigned)xq >= 128u) continue;
      acc += W[o * 9 + ky * 3 + kx] * (float)p[yq * 128 + xq];
    }
  }
  Z[(((size_t)(s * 1152 + o)) << 14) + n] = (_Float16)acc;
}

// ---------------------------------------------------------------------------
// 1/L2-norm over n=16384 for the q (o<384) and k (384<=o<768) rows of Z.
// One block per row; rn[s*768+o] = 1/max(||row||, 1e-12)
// ---------------------------------------------------------------------------
__global__ __launch_bounds__(256)
void row_rnorm(const _Float16* __restrict__ Z, float* __restrict__ rn) {
  __shared__ float red[256];
  int row = blockIdx.x;              // 0..6143
  int s = row / 768, o = row % 768;
  const _Float16* p = Z + ((size_t)(s * 1152 + o) << 14);
  float acc = 0.f;
  for (int i = threadIdx.x; i < 16384; i += 256) {
    float v = (float)p[i];
    acc += v * v;
  }
  red[threadIdx.x] = acc;
  __syncthreads();
  for (int st = 128; st > 0; st >>= 1) {
    if ((int)threadIdx.x < st) red[threadIdx.x] += red[threadIdx.x + st];
    __syncthreads();
  }
  if (threadIdx.x == 0) rn[row] = 1.0f / fmaxf(sqrtf(red[0]), 1e-12f);
}

// ---------------------------------------------------------------------------
// Fused channel attention per (head, batch):
//  S = Qcat(96 x 16384) * Kc^T, scaled by rq*rk*temperature, softmax over 96,
//  OUT = P * Vc (96 x 16384) -> Cat[b][768][16384] in '(head c)' order.
// Block = 192 threads = 6 waves; wave w owns 16-row strip w.
// ---------------------------------------------------------------------------
__global__ __launch_bounds__(192)
void channel_attention(const _Float16* __restrict__ Z, const float* __restrict__ rn,
                       const float* __restrict__ temperature, _Float16* __restrict__ Cat) {
  __shared__ __align__(32) float S[96 * 96];           // 36864 B
  __shared__ __align__(32) _Float16 P[96 * 104];       // 19968 B (padded rows)
  _Float16* VcT = (_Float16*)S;                        // phase-3 reuse: 128*104*2 = 26624 B

  const int h = blockIdx.x, b = blockIdx.y;
  const int tid = threadIdx.x, lane = tid & 31, w = tid >> 5;  // w in 0..5
  const int la = lane & 15, lh = lane >> 4;
  const int kA = lh * 16, kB = lh * 16;

  const _Float16* Zx = Z + (((size_t)b * 1152) << 14);
  const _Float16* Zm = Z + (((size_t)(4 + b) * 1152) << 14);
  const float* rnx = rn + b * 768;
  const float* rnm = rn + (4 + b) * 768;
  const float temp = temperature[h];

  // ---- Phase 1: S = Qcat * Kc^T (K = 16384), direct global WMMA fragments ----
  const int i = 16 * w + la;  // logical row 0..95 (0..47 = q, 48..95 = q_m)
  const _Float16* Arow = (i < 48) ? (Zx + ((size_t)(h * 48 + i) << 14))
                                  : (Zm + ((size_t)(h * 48 + i - 48) << 14));
  const _Float16* Brow[6];
#pragma unroll
  for (int nj = 0; nj < 6; ++nj) {
    int j = 16 * nj + la;     // 0..47 = k, 48..95 = k_m
    Brow[nj] = (j < 48) ? (Zx + ((size_t)(384 + h * 48 + j) << 14))
                        : (Zm + ((size_t)(384 + h * 48 + j - 48) << 14));
  }
  v8f acc[6];
#pragma unroll
  for (int nj = 0; nj < 6; ++nj) acc[nj] = vzero();

  for (int n0 = 0; n0 < 16384; n0 += 32) {
    V16 a;
    a.h[0] = *(const v8h*)(Arow + n0 + kA);
    a.h[1] = *(const v8h*)(Arow + n0 + kA + 8);
#pragma unroll
    for (int nj = 0; nj < 6; ++nj) {
      v16h bb = *(const v16h*)(Brow[nj] + n0 + kB);
      acc[nj] = wmma_f16(a.v, bb, acc[nj]);
    }
  }
#pragma unroll
  for (int nj = 0; nj < 6; ++nj) {
    int col = 16 * nj + la;
    float rk = (col < 48) ? rnx[384 + h * 48 + col] : rnm[384 + h * 48 + col - 48];
#pragma unroll
    for (int r = 0; r < 8; ++r) {
      int row = 16 * w + r + lh * 8;
      float rq = (row < 48) ? rnx[h * 48 + row] : rnm[h * 48 + row - 48];
      S[row * 96 + col] = acc[nj][r] * rq * rk * temp;
    }
  }
  __syncthreads();

  // ---- Phase 2: softmax over 96 cols, one thread per row, P in f16 ----
  if (tid < 96) {
    float m = -1e30f;
    for (int j = 0; j < 96; ++j) m = fmaxf(m, S[tid * 96 + j]);
    float sum = 0.f;
    for (int j = 0; j < 96; ++j) sum += __expf(S[tid * 96 + j] - m);
    float inv = 1.0f / sum;
    for (int j = 0; j < 96; ++j)
      P[tid * 104 + j] = (_Float16)(__expf(S[tid * 96 + j] - m) * inv);
  }
  __syncthreads();

  // ---- Phase 3: OUT = P * Vc ----
  V16 ap[3];
#pragma unroll
  for (int ks = 0; ks < 3; ++ks) {
    ap[ks].h[0] = *(const v8h*)&P[i * 104 + ks * 32 + kA];
    ap[ks].h[1] = *(const v8h*)&P[i * 104 + ks * 32 + kA + 8];
  }
  const int jst = tid >> 1;             // 0..95: V row to stage
  const int ncst = (tid & 1) * 64;      // which half of 128 columns
  const _Float16* Vrow = (jst < 48) ? (Zx + ((size_t)(768 + h * 48 + jst) << 14))
                                    : (Zm + ((size_t)(768 + h * 48 + jst - 48) << 14));
  for (int n0 = 0; n0 < 16384; n0 += 128) {
    // stage VcT[n][j] (transposed) in LDS
#pragma unroll
    for (int c8 = 0; c8 < 8; ++c8) {
      V8 t; t.v = *(const v8h*)(Vrow + n0 + ncst + c8 * 8);
#pragma unroll
      for (int e = 0; e < 8; ++e) VcT[(ncst + c8 * 8 + e) * 104 + jst] = t.e[e];
    }
    __syncthreads();
#pragma unroll
    for (int ns = 0; ns < 8; ++ns) {
      int n = 16 * ns + la;
      v8f o = vzero();
#pragma unroll
      for (int ks = 0; ks < 3; ++ks) {
        v16h bb = *(const v16h*)&VcT[n * 104 + ks * 32 + kB];
        o = wmma_f16(ap[ks].v, bb, o);
      }
#pragma unroll
      for (int r = 0; r < 8; ++r) {
        int row = 16 * w + r + lh * 8;
        int catRow = (row < 48) ? (h * 48 + row) : (384 + h * 48 + row - 48);
        Cat[((size_t)b * 768 + catRow) * 16384 + n0 + 16 * ns + la] = (_Float16)o[r];
      }
    }
    __syncthreads();
  }
}

// ---------------------------------------------------------------------------
extern "C" void kernel_launch(void* const* d_in, const int* in_sizes, int n_in,
                              void* d_out, int out_size, void* d_ws, size_t ws_size,
                              hipStream_t stream) {
  (void)in_sizes; (void)n_in; (void)out_size; (void)ws_size;
  const float* x     = (const float*)d_in[0];
  const float* msg   = (const float*)d_in[1];
  const float* temp  = (const float*)d_in[2];
  const float* qkvw  = (const float*)d_in[3];
  const float* dww   = (const float*)d_in[4];
  const float* projw = (const float*)d_in[5];
  const float* linw  = (const float*)d_in[6];
  const float* linb  = (const float*)d_in[7];

  char* ws = (char*)d_ws;
  size_t off = 0;
  auto alloc = [&](size_t bytes) -> void* {
    void* p = ws + off;
    off = (off + bytes + 255) & ~(size_t)255;
    return p;
  };

  const size_t N = 16384;  // 128*128
  _Float16* INh = (_Float16*)alloc((size_t)8 * 384 * N * 2);
  _Float16* Wqh = (_Float16*)alloc((size_t)1152 * 384 * 2);
  _Float16* Lih = (_Float16*)alloc((size_t)384 * 768 * 2);
  _Float16* Prh = (_Float16*)alloc((size_t)384 * 384 * 2);
  _Float16* Yh  = (_Float16*)alloc((size_t)8 * 1152 * N * 2);
  _Float16* Zh  = (_Float16*)alloc((size_t)8 * 1152 * N * 2);
  float*    rnm = (float*)   alloc((size_t)8 * 768 * 4);
  _Float16* Cat = (_Float16*)alloc((size_t)4 * 768 * N * 2);
  _Float16* Rb  = (_Float16*)alloc((size_t)4 * 384 * N * 2);

  // 0) fp32 -> f16 conversions
  convert_f32_f16<<<24576, 256, 0, stream>>>(x,   INh,                         6291456);
  convert_f32_f16<<<24576, 256, 0, stream>>>(msg, INh + (size_t)4 * 384 * N,   6291456);
  convert_f32_f16<<<  432, 256, 0, stream>>>(qkvw, Wqh, 110592);
  convert_f32_f16<<<  288, 256, 0, stream>>>(linw, Lih,  73728);
  convert_f32_f16<<<  144, 256, 0, stream>>>(projw, Prh, 36864);

  // 1) QKV pointwise GEMM: Y[s] = qkv_w(1152x384) * IN[s](384x16384), s=0..7
  gemm_f16<_Float16, false><<<dim3(128, 9, 8), 256, 0, stream>>>(
      Wqh, INh, Yh, nullptr, 1152, 16384, 384,
      (size_t)0, (size_t)384 * N, (size_t)1152 * N);

  // 2) depthwise 3x3 SAME
  dwconv3x3<<<dim3(64, 1152, 8), 256, 0, stream>>>(Yh, dww, Zh);

  // 3) reciprocal L2 norms of q/k rows
  row_rnorm<<<6144, 256, 0, stream>>>(Zh, rnm);

  // 4) fused channel attention -> Cat[b][768][16384]
  channel_attention<<<dim3(8, 4), 192, 0, stream>>>(Zh, rnm, temp, Cat);

  // 5) Linear: R[b] = lin_w(384x768) * Cat[b](768x16384) + lin_b
  gemm_f16<_Float16, true><<<dim3(128, 3, 4), 256, 0, stream>>>(
      Lih, Cat, Rb, linb, 384, 16384, 768,
      (size_t)0, (size_t)768 * N, (size_t)384 * N);

  // 6) Projection: out[b] = proj_w(384x384) * R[b] -> fp32 d_out
  gemm_f16<float, false><<<dim3(128, 3, 4), 256, 0, stream>>>(
      Prh, Rb, (float*)d_out, nullptr, 384, 16384, 384,
      (size_t)0, (size_t)384 * N, (size_t)384 * N);
}
